// Mamba_20392504721948
// MI455X (gfx1250) — matrix operand order repaired and checked by reading, not compiled, not run on hardware
//
#include <hip/hip_runtime.h>
#include <hip/hip_bf16.h>
#include <math.h>

// ---------------------------------------------------------------------------
// Model dims
// ---------------------------------------------------------------------------
#define NB   16      // batch
#define TT   20      // time steps
#define VR_  30      // joints after VORDER gather
#define JC_  64
#define DM_  1280    // mamba model dim = TT*JC
#define TL_  1920    // VR*JC
#define TE_  512
#define DIN_ 2560
#define DST_ 16
#define DTR_ 80
#define NH_  8
#define HD_  240
#define FFN_ 2560
#define RM_  (NB*VR_)   // 480 mamba rows
#define RT_  (NB*TT)    // 320 temporal rows
#define HELEMS (NB*TT*VR_*JC_)  // 614400

__constant__ int c_vorder[30] = {9,8,7,10,11,12,11,10,7,13,14,15,14,13,7,6,
                                 0,1,2,1,0,6,3,4,5,4,3,6,7,8};
__constant__ int c_rev[16]    = {20,19,18,26,25,24,27,28,29,0,7,6,5,13,12,11};

// ---------------------------------------------------------------------------
// WMMA types & helpers
// ---------------------------------------------------------------------------
typedef __attribute__((ext_vector_type(16))) __bf16          v16bf;
typedef __attribute__((ext_vector_type(16))) unsigned short  v16us;
typedef __attribute__((ext_vector_type(8)))  float           v8f;

// fp32 -> bf16, round-half-up (cheap; 0.5 ulp of bf16 vs RNE)
__device__ __forceinline__ unsigned short f2bf(float f) {
    return (unsigned short)((__builtin_bit_cast(unsigned, f) + 0x8000u) >> 16);
}
// pack two floats into one dword of bf16: 2x v_add + 1x v_perm_b32
__device__ __forceinline__ unsigned pack_bf16(float lo, float hi) {
    unsigned ul = __builtin_bit_cast(unsigned, lo) + 0x8000u;
    unsigned uh = __builtin_bit_cast(unsigned, hi) + 0x8000u;
#if __has_builtin(__builtin_amdgcn_perm)
    return __builtin_amdgcn_perm(uh, ul, 0x07060302u);   // {uh.b3,uh.b2,ul.b3,ul.b2}
#else
    return (uh & 0xffff0000u) | (ul >> 16);
#endif
}
__device__ __forceinline__ float act_silu(float x) { return x / (1.f + __expf(-x)); }
__device__ __forceinline__ float act_gelu(float x) {
    float x3 = x * x * x;
    return 0.5f * x * (1.f + tanhf(0.7978845608f * (x + 0.044715f * x3)));
}
__device__ __forceinline__ float act_softplus(float x) {
    return (x > 20.f) ? x : log1pf(__expf(x));
}

// ---------------------------------------------------------------------------
// Generic WMMA GEMM:  C = [res +] act( A(MxK) @ W(KxN) [+ bias] )
// 64x64 block tile, K unrolled 2x32 per barrier round; 8 waves as 4(M)x2(N).
// LDS holds bf16 fragments already in WMMA register layout ([frag][lane][e]):
//   A frag: lane = (m&15) + 16*(k>>4),     e = k&15
//   B frag: lane = (n&15) + 16*(k>>4),     e = k&15
// so operand fetch is one aligned v16us LDS load per fragment.
// All staging global loads are single b128 per thread (no address chaining).
// act: 0=none 1=gelu 2=softplus 3=silu
// ---------------------------------------------------------------------------
__global__ __launch_bounds__(256)
void k_gemm(const float* __restrict__ A, int lda,
            const float* __restrict__ W, int ldw,
            const float* __restrict__ bias,
            const float* __restrict__ res, int ldr,
            float* __restrict__ C, int ldc,
            int M, int N, int K, int act)
{
    __shared__ __align__(32) unsigned short aF[2][4][32][16]; // [slab][m-grp][lane][e]
    __shared__ __align__(32) unsigned short bF[2][4][32][16]; // [slab][n-grp][lane][e]

    const int tid  = threadIdx.x;
    const int lane = tid & 31;
    const int wave = tid >> 5;
    const int wm   = wave >> 1;          // 0..3
    const int wn   = wave & 1;           // 0..1
    const int bm   = blockIdx.y << 6;
    const int bn   = blockIdx.x << 6;
    const int fm   = lane & 15;
    const int fh   = lane >> 4;

    const bool full = (bm + 64 <= M) && (bn + 64 <= N);

    // uniform bases (SGPR); per-lane offsets stay 32-bit unsigned -> GVS mode
    const float* __restrict__ Ab = A + (size_t)bm * lda;
    const float* __restrict__ Wb = W + bn;

    v8f acc0 = {}; v8f acc1 = {};

    // stage one 64x32 A tile + 32x64 W tile into slab s (unguarded, vectorized)
    auto stage_fast = [&](int kk, int s) {
        #pragma unroll
        for (int it = 0; it < 2; ++it) {
            int idx = tid + (it << 8);                  // 0..511
            {   // A: row-major float4 along K (contiguous) -> packed b64 store
                int row = idx >> 3;                     // 0..63
                int c4  = (idx & 7) << 2;               // 0..28
                unsigned aoff = (unsigned)(row * lda + kk + c4);
                const float4 v = *(const float4*)(Ab + aoff);
                int half = c4 >> 4;
                int e0   = c4 & 15;
                unsigned short* dst = &aF[s][row >> 4][(row & 15) + (half << 4)][e0];
                uint2 p;
                p.x = pack_bf16(v.x, v.y);
                p.y = pack_bf16(v.z, v.w);
                *(uint2*)dst = p;
            }
            {   // W: contiguous float4 along N -> 4 b16 scatter stores to
                // consecutive lanes (+32B apart: immediate-offset ds stores)
                int r  = idx >> 4;                      // 0..31 (K within slab)
                int n4 = (idx & 15) << 2;               // 0..60
                unsigned woff = (unsigned)((kk + r) * ldw + n4);
                const float4 v = *(const float4*)(Wb + woff);
                int e  = r & 15;
                int lb = (n4 & 15) + ((r >> 4) << 4);   // lane base (n4&15 in 0,4,8,12)
                int jg = n4 >> 4;
                bF[s][jg][lb + 0][e] = f2bf(v.x);
                bF[s][jg][lb + 1][e] = f2bf(v.y);
                bF[s][jg][lb + 2][e] = f2bf(v.z);
                bF[s][jg][lb + 3][e] = f2bf(v.w);
            }
        }
    };

    // guarded staging into slab 0 (branchless clamp + select)
    auto stage_edge = [&](int kk) {
        #pragma unroll
        for (int i = 0; i < 8; ++i) {
            int idx = tid + (i << 8);
            int row = idx >> 5, k = idx & 31;
            int gr = bm + row, gc = kk + k;
            int cr = gr < M ? gr : M - 1;
            int cc = gc < K ? gc : K - 1;
            float v = A[cr * lda + cc];
            if (gr >= M || gc >= K) v = 0.f;
            aF[0][row >> 4][(row & 15) + ((k >> 4) << 4)]
              [k & 15] = f2bf(v);
        }
        #pragma unroll
        for (int i = 0; i < 8; ++i) {
            int idx = tid + (i << 8);
            int k = idx >> 6, n = idx & 63;
            int gr = kk + k, gc = bn + n;
            int cr = gr < K ? gr : K - 1;
            int cc = gc < N ? gc : N - 1;
            float v = W[cr * ldw + cc];
            if (gr >= K || gc >= N) v = 0.f;
            bF[0][n >> 4][(n & 15) + ((k >> 4) << 4)][k & 15] = f2bf(v);
        }
    };

    // one 16x16x32 MAC round from slab s
    auto mma = [&](int s) {
        v16bf a  = __builtin_bit_cast(v16bf, *(const v16us*)aF[s][wm][lane]);
        v16bf b0 = __builtin_bit_cast(v16bf, *(const v16us*)bF[s][(wn << 1) + 0][lane]);
        v16bf b1 = __builtin_bit_cast(v16bf, *(const v16us*)bF[s][(wn << 1) + 1][lane]);
        acc0 = __builtin_amdgcn_wmma_f32_16x16x32_bf16(false, a, false, b0,
                                                       (short)0, acc0, false, false);
        acc1 = __builtin_amdgcn_wmma_f32_16x16x32_bf16(false, a, false, b1,
                                                       (short)0, acc1, false, false);
    };

    int k0 = 0;
    if (full) {
        for (; k0 + 64 <= K; k0 += 64) {
            stage_fast(k0, 0);
            stage_fast(k0 + 32, 1);
            __syncthreads();
            if (tid < 32) {                             // prefetch next slab to GL2
                int pr = k0 + 64 + tid;
                if (pr < K) __builtin_prefetch(&W[(size_t)pr * ldw + bn], 0, 1);
                int ar = bm + (tid << 1);
                if (k0 + 64 < K && ar < M)
                    __builtin_prefetch(&A[(size_t)ar * lda + k0 + 64], 0, 1);
            }
            mma(0);
            mma(1);
            __syncthreads();
        }
    }
    for (; k0 < K; k0 += 32) {                          // tail / edge tiles
        stage_edge(k0);
        __syncthreads();
        mma(0);
        __syncthreads();
    }

    // ---- epilogue + store (C/D layout: row = r + 8*half, col = fm) ----
    const int col0 = bn + (wn << 5) + fm;
    const int col1 = col0 + 16;
    const float bi0 = (bias && col0 < N) ? bias[col0] : 0.f;
    const float bi1 = (bias && col1 < N) ? bias[col1] : 0.f;
    #pragma unroll
    for (int r = 0; r < 8; ++r) {
        int row = bm + (wm << 4) + r + (fh << 3);
        if (row >= M) continue;
        if (col0 < N) {
            float v = acc0[r] + bi0;
            if      (act == 1) v = act_gelu(v);
            else if (act == 2) v = act_softplus(v);
            else if (act == 3) v = act_silu(v);
            if (res) v += res[row * ldr + col0];
            C[row * ldc + col0] = v;
        }
        if (col1 < N) {
            float v = acc1[r] + bi1;
            if      (act == 1) v = act_gelu(v);
            else if (act == 2) v = act_softplus(v);
            else if (act == 3) v = act_silu(v);
            if (res) v += res[row * ldr + col1];
            C[row * ldc + col1] = v;
        }
    }
}

// ---------------------------------------------------------------------------
// Small fused kernels
// ---------------------------------------------------------------------------
__global__ void k_time_embed(const float* __restrict__ ts, float* __restrict__ out) {
    int i = blockIdx.x * 256 + threadIdx.x;            // NB*640
    if (i >= NB * 640) return;
    int b = i / 640, j = i % 640;
    float f = __expf(-logf(10000.f) * (float)j / 640.f);
    float a = ts[b] * f;
    out[b * DM_ + j]       = cosf(a);
    out[b * DM_ + 640 + j] = sinf(a);
}

__global__ void k_silu_vec(const float* __restrict__ in, float* __restrict__ out, int n) {
    int i = blockIdx.x * 256 + threadIdx.x;
    if (i < n) out[i] = act_silu(in[i]);
}

__global__ void k_embed(const float* __restrict__ x, const float* __restrict__ ew,
                        const float* __restrict__ eb, const float* __restrict__ seq,
                        float* __restrict__ h) {
    int i = blockIdx.x * 256 + threadIdx.x;            // HELEMS
    if (i >= HELEMS) return;
    int c = i & 63;
    int v = (i >> 6) % VR_;
    int t = (i / (64 * VR_)) % TT;
    int b = i / (64 * VR_ * TT);
    int j = c_vorder[v];
    const float* xp = &x[(b * TT + t) * 48 + j * 3];
    float s = eb[c] + seq[(t * 16 + j) * 64 + c];
    s += xp[0] * ew[c] + xp[1] * ew[64 + c] + xp[2] * ew[128 + c];
    h[i] = s;
}

// RMSNorm: out = x * rsqrt(mean(x^2)+eps) * w  [+ embp[row/rows_per_b]]
__global__ __launch_bounds__(256)
void k_rms(const float* __restrict__ in, const float* __restrict__ w,
           const float* __restrict__ embp, float* __restrict__ out,
           int D, int rows_per_b) {
    __shared__ float red[256];
    int row = blockIdx.x, tid = threadIdx.x;
    const float* xr = in + row * D;
    float s = 0.f;
    for (int d = tid; d < D; d += 256) { float v = xr[d]; s += v * v; }
    red[tid] = s; __syncthreads();
    for (int o = 128; o > 0; o >>= 1) { if (tid < o) red[tid] += red[tid + o]; __syncthreads(); }
    float sc = rsqrtf(red[0] / (float)D + 1e-5f);
    int b = row / rows_per_b;
    for (int d = tid; d < D; d += 256) {
        float v = xr[d] * sc * w[d];
        if (embp) v += embp[b * D + d];
        out[row * D + d] = v;
    }
}

// LayerNorm: out = (x-m)*rsqrt(var+eps)*s + b [+ embp[row/rows_per_b]]
__global__ __launch_bounds__(256)
void k_ln(const float* __restrict__ in, const float* __restrict__ sw,
          const float* __restrict__ sb, const float* __restrict__ embp,
          float* __restrict__ out, int D, int rows_per_b) {
    __shared__ float r1[256], r2[256];
    int row = blockIdx.x, tid = threadIdx.x;
    const float* xr = in + row * D;
    float s = 0.f, q = 0.f;
    for (int d = tid; d < D; d += 256) { float v = xr[d]; s += v; q += v * v; }
    r1[tid] = s; r2[tid] = q; __syncthreads();
    for (int o = 128; o > 0; o >>= 1) {
        if (tid < o) { r1[tid] += r1[tid + o]; r2[tid] += r2[tid + o]; }
        __syncthreads();
    }
    float m  = r1[0] / (float)D;
    float vr = r2[0] / (float)D - m * m;
    float sc = rsqrtf(vr + 1e-5f);
    int b = row / rows_per_b;
    for (int d = tid; d < D; d += 256) {
        float v = (xr[d] - m) * sc * sw[d] + sb[d];
        if (embp) v += embp[b * D + d];
        out[row * D + d] = v;
    }
}

// causal depthwise conv (k=4) over L=30 + bias + SiLU; input is xr[:, :DIN]
__global__ void k_dwconv_silu(const float* __restrict__ xr, const float* __restrict__ cw,
                              const float* __restrict__ cb, float* __restrict__ xc) {
    int i = blockIdx.x * 256 + threadIdx.x;            // RM_*DIN_
    if (i >= RM_ * DIN_) return;
    int d = i % DIN_, row = i / DIN_;
    int l = row % VR_, b = row / VR_;
    float acc = cb[d];
    #pragma unroll
    for (int j = 0; j < 4; ++j) {
        int ls = l - 3 + j;
        if (ls >= 0) acc += xr[(b * VR_ + ls) * (2 * DIN_) + d] * cw[d * 4 + j];
    }
    xc[i] = act_silu(acc);
}

// selective scan, fused with (+u*D) * silu(res)
__global__ void k_scan(const float* __restrict__ xc, const float* __restrict__ dlt,
                       const float* __restrict__ xdbl, const float* __restrict__ A_log,
                       const float* __restrict__ Dp, const float* __restrict__ xr,
                       float* __restrict__ yz) {
    int i = blockIdx.x * 256 + threadIdx.x;            // NB*DIN_
    if (i >= NB * DIN_) return;
    int d = i % DIN_, b = i / DIN_;
    float A[DST_], s[DST_];
    #pragma unroll
    for (int n = 0; n < DST_; ++n) { A[n] = -__expf(A_log[d * DST_ + n]); s[n] = 0.f; }
    float Dv = Dp[d];
    for (int l = 0; l < VR_; ++l) {
        int row = b * VR_ + l;
        float dt = dlt[row * DIN_ + d];
        float u  = xc[row * DIN_ + d];
        const float* Bm = &xdbl[row * 112 + DTR_];
        const float* Cm = &xdbl[row * 112 + DTR_ + DST_];
        float y = 0.f;
        #pragma unroll
        for (int n = 0; n < DST_; ++n) {
            s[n] = s[n] * __expf(dt * A[n]) + dt * Bm[n] * u;
            y += s[n] * Cm[n];
        }
        float res = xr[row * (2 * DIN_) + DIN_ + d];
        yz[row * DIN_ + d] = (y + u * Dv) * act_silu(res);
    }
}

// attention: one block per (b, head, query); L=20, HD=240
__global__ __launch_bounds__(256)
void k_attn(const float* __restrict__ qkv, float* __restrict__ attn) {
    int blk = blockIdx.x;                              // NB*NH_*TT
    int tq = blk % TT, h = (blk / TT) % NH_, b = blk / (TT * NH_);
    __shared__ float qs[HD_], sc[TT], ps[TT], red[1];
    const float* qrow = &qkv[(b * TT + tq) * (3 * TL_) + h * HD_];
    for (int d = threadIdx.x; d < HD_; d += 256) qs[d] = qrow[d];
    __syncthreads();
    if (threadIdx.x < TT) {
        int k = threadIdx.x;
        const float* krow = &qkv[(b * TT + k) * (3 * TL_) + TL_ + h * HD_];
        float ds = 0.f;
        for (int d = 0; d < HD_; ++d) ds += qs[d] * krow[d];
        sc[k] = ds * 0.06454972243679028f;             // 1/sqrt(240)
    }
    __syncthreads();
    if (threadIdx.x == 0) {
        float mx = sc[0];
        for (int k = 1; k < TT; ++k) mx = fmaxf(mx, sc[k]);
        float den = 0.f;
        for (int k = 0; k < TT; ++k) { float e = __expf(sc[k] - mx); ps[k] = e; den += e; }
        red[0] = 1.f / den;
    }
    __syncthreads();
    float inv = red[0];
    for (int d = threadIdx.x; d < HD_; d += 256) {
        float a = 0.f;
        for (int k = 0; k < TT; ++k)
            a += ps[k] * qkv[(b * TT + k) * (3 * TL_) + 2 * TL_ + h * HD_ + d];
        attn[(b * TT + tq) * TL_ + h * HD_ + d] = a * inv;
    }
}

// (B,T,VR,JC) -> (B,VR, T*JC)
__global__ void k_h2v(const float* __restrict__ h, float* __restrict__ hv) {
    int i = blockIdx.x * 256 + threadIdx.x;
    if (i >= HELEMS) return;
    int c = i & 63;
    int t = (i >> 6) % TT;
    int row = i / DM_;
    int v = row % VR_, b = row / VR_;
    hv[i] = h[((b * TT + t) * VR_ + v) * 64 + c];
}

// (B,VR,T*JC) -> (B,T, VR*JC)
__global__ void k_v2t(const float* __restrict__ hv, float* __restrict__ ht) {
    int i = blockIdx.x * 256 + threadIdx.x;
    if (i >= HELEMS) return;
    int c = i & 63;
    int v = (i >> 6) % VR_;
    int t = (i / TL_) % TT;
    int b = i / (TL_ * TT);
    ht[i] = hv[(b * VR_ + v) * DM_ + t * 64 + c];
}

// elementwise out = a + (b? b : 0)   (note (B,T,VR,JC) flat == (B,T,TL) flat)
__global__ void k_add2(const float* __restrict__ a, const float* __restrict__ b_,
                       float* __restrict__ o, int n) {
    int i = blockIdx.x * 256 + threadIdx.x;
    if (i < n) o[i] = a[i] + (b_ ? b_[i] : 0.f);
}

// final head: out[b,t,j,o] = sum_c hn[b,t, rev[j]*64+c] * lm_w[c,o] + lm_b[o]
__global__ void k_head(const float* __restrict__ hn, const float* __restrict__ lw,
                       const float* __restrict__ lb, float* __restrict__ out) {
    int i = blockIdx.x * 256 + threadIdx.x;            // NB*TT*16*3
    if (i >= NB * TT * 16 * 3) return;
    int o = i % 3, j = (i / 3) % 16, t = (i / 48) % TT, b = i / (48 * TT);
    int v = c_rev[j];
    const float* hr = &hn[(b * TT + t) * TL_ + v * 64];
    float s = lb[o];
    for (int c = 0; c < 64; ++c) s += hr[c] * lw[c * 3 + o];
    out[i] = s;
}

// ---------------------------------------------------------------------------
// Host orchestration
// ---------------------------------------------------------------------------
static inline void launch_gemm(hipStream_t st, const float* A, int lda,
                               const float* W, int ldw, const float* bias,
                               const float* res, int ldr, float* C, int ldc,
                               int M, int N, int K, int act) {
    dim3 g((N + 63) / 64, (M + 63) / 64);
    k_gemm<<<g, 256, 0, st>>>(A, lda, W, ldw, bias, res, ldr, C, ldc, M, N, K, act);
}

extern "C" void kernel_launch(void* const* d_in, const int* in_sizes, int n_in,
                              void* d_out, int out_size, void* d_ws, size_t ws_size,
                              hipStream_t stream) {
    const float* p_x       = (const float*)d_in[0];
    const float* p_ts      = (const float*)d_in[1];
    const float* p_emb_w   = (const float*)d_in[2];
    const float* p_emb_b   = (const float*)d_in[3];
    const float* p_seq     = (const float*)d_in[4];
    const float* p_tw1     = (const float*)d_in[5];
    const float* p_tb1     = (const float*)d_in[6];
    const float* p_tw2     = (const float*)d_in[7];
    const float* p_tb2     = (const float*)d_in[8];
    const float* p_mrms    = (const float*)d_in[9];
    const float* p_membw   = (const float*)d_in[10];
    const float* p_membb   = (const float*)d_in[11];
    const float* p_minw    = (const float*)d_in[12];
    const float* p_mconvw  = (const float*)d_in[13];
    const float* p_mconvb  = (const float*)d_in[14];
    const float* p_mxproj  = (const float*)d_in[15];
    const float* p_mdtw    = (const float*)d_in[16];
    const float* p_mdtb    = (const float*)d_in[17];
    const float* p_mAlog   = (const float*)d_in[18];
    const float* p_mD      = (const float*)d_in[19];
    const float* p_moutw   = (const float*)d_in[20];
    const float* p_ln1s    = (const float*)d_in[21];
    const float* p_ln1b    = (const float*)d_in[22];
    const float* p_ln2s    = (const float*)d_in[23];
    const float* p_ln2b    = (const float*)d_in[24];
    const float* p_tembw   = (const float*)d_in[25];
    const float* p_tembb   = (const float*)d_in[26];
    const float* p_qkvw    = (const float*)d_in[27];
    const float* p_qkvb    = (const float*)d_in[28];
    const float* p_toutw   = (const float*)d_in[29];
    const float* p_toutb   = (const float*)d_in[30];
    const float* p_w1      = (const float*)d_in[31];
    const float* p_b1      = (const float*)d_in[32];
    const float* p_w2      = (const float*)d_in[33];
    const float* p_b2      = (const float*)d_in[34];
    const float* p_normf   = (const float*)d_in[35];
    const float* p_lmw     = (const float*)d_in[36];
    const float* p_lmb     = (const float*)d_in[37];

    (void)in_sizes; (void)n_in; (void)out_size; (void)ws_size;

    float* wp = (float*)d_ws;
    auto alloc = [&](size_t n) { float* r = wp; wp += n; return r; };
    float* temb  = alloc((size_t)NB * DM_);
    float* e1    = alloc((size_t)NB * TE_);
    float* embv  = alloc((size_t)NB * TE_);
    float* sembv = alloc((size_t)NB * TE_);
    float* membp = alloc((size_t)NB * DM_);
    float* tembp = alloc((size_t)NB * TL_);
    float* hbuf  = alloc(HELEMS);
    float* save0 = alloc(HELEMS);
    float* save1 = alloc(HELEMS);
    float* save2 = alloc(HELEMS);
    float* save3 = alloc(HELEMS);
    float* saves[4] = {save0, save1, save2, save3};
    float* hv    = alloc(HELEMS);
    float* hn    = alloc(HELEMS);
    float* xr    = alloc((size_t)RM_ * 2 * DIN_);
    float* xc    = alloc((size_t)RM_ * DIN_);
    float* xdbl  = alloc((size_t)RM_ * 112);
    float* dlt   = alloc((size_t)RM_ * DIN_);
    float* yz    = alloc((size_t)RM_ * DIN_);
    float* ht    = alloc((size_t)RT_ * TL_);
    float* ybuf  = alloc((size_t)RT_ * TL_);
    float* qkvb  = alloc((size_t)RT_ * 3 * TL_);
    float* attn  = alloc((size_t)RT_ * TL_);
    float* zbuf  = alloc((size_t)RT_ * TL_);
    float* znbuf = alloc((size_t)RT_ * TL_);
    float* f1    = alloc((size_t)RT_ * FFN_);

    const int GH = (HELEMS + 255) / 256;

    // ---- time embedding MLP ----
    k_time_embed<<<(NB * 640 + 255) / 256, 256, 0, stream>>>(p_ts, temb);
    launch_gemm(stream, temb, DM_, p_tw1, TE_, p_tb1, nullptr, 0, e1, TE_,
                NB, TE_, DM_, 3);                                     // silu
    launch_gemm(stream, e1, TE_, p_tw2, TE_, p_tb2, nullptr, 0, embv, TE_,
                NB, TE_, TE_, 0);
    k_silu_vec<<<(NB * TE_ + 255) / 256, 256, 0, stream>>>(embv, sembv, NB * TE_);

    // ---- input embedding + VORDER gather ----
    k_embed<<<GH, 256, 0, stream>>>(p_x, p_emb_w, p_emb_b, p_seq, hbuf);

    for (int i = 0; i < 8; ++i) {
        // per-layer emb projections (M=16 GEMMs)
        launch_gemm(stream, sembv, TE_, p_membw + (size_t)i * TE_ * DM_, DM_,
                    p_membb + (size_t)i * DM_, nullptr, 0, membp, DM_,
                    NB, DM_, TE_, 0);
        launch_gemm(stream, sembv, TE_, p_tembw + (size_t)i * TE_ * TL_, TL_,
                    p_tembb + (size_t)i * TL_, nullptr, 0, tembp, TL_,
                    NB, TL_, TE_, 0);

        if (i < 4)
            hipMemcpyAsync(saves[i], hbuf, (size_t)HELEMS * sizeof(float),
                           hipMemcpyDeviceToDevice, stream);

        // ---- Mamba branch on (B,VR, T*JC) ----
        k_h2v<<<GH, 256, 0, stream>>>(hbuf, hv);
        k_rms<<<RM_, 256, 0, stream>>>(hv, p_mrms + (size_t)i * DM_, membp, hn, DM_, VR_);
        launch_gemm(stream, hn, DM_, p_minw + (size_t)i * DM_ * 2 * DIN_, 2 * DIN_,
                    nullptr, nullptr, 0, xr, 2 * DIN_, RM_, 2 * DIN_, DM_, 0);
        k_dwconv_silu<<<(RM_ * DIN_ + 255) / 256, 256, 0, stream>>>(
            xr, p_mconvw + (size_t)i * DIN_ * 4, p_mconvb + (size_t)i * DIN_, xc);
        launch_gemm(stream, xc, DIN_, p_mxproj + (size_t)i * DIN_ * 112, 112,
                    nullptr, nullptr, 0, xdbl, 112, RM_, 112, DIN_, 0);
        launch_gemm(stream, xdbl, 112, p_mdtw + (size_t)i * DTR_ * DIN_, DIN_,
                    p_mdtb + (size_t)i * DIN_, nullptr, 0, dlt, DIN_,
                    RM_, DIN_, DTR_, 2);                              // softplus
        k_scan<<<(NB * DIN_ + 255) / 256, 256, 0, stream>>>(
            xc, dlt, xdbl, p_mAlog + (size_t)i * DIN_ * DST_,
            p_mD + (size_t)i * DIN_, xr, yz);
        launch_gemm(stream, yz, DIN_, p_moutw + (size_t)i * DIN_ * DM_, DM_,
                    nullptr, hv, DM_, hv, DM_, RM_, DM_, DIN_, 0);    // hv += out

        // ---- temporal attention on (B,T, VR*JC) ----
        k_v2t<<<GH, 256, 0, stream>>>(hv, ht);
        k_ln<<<RT_, 256, 0, stream>>>(ht, p_ln1s + (size_t)i * TL_,
                                      p_ln1b + (size_t)i * TL_, tembp, ybuf, TL_, TT);
        launch_gemm(stream, ybuf, TL_, p_qkvw + (size_t)i * TL_ * 3 * TL_, 3 * TL_,
                    p_qkvb + (size_t)i * 3 * TL_, nullptr, 0, qkvb, 3 * TL_,
                    RT_, 3 * TL_, TL_, 0);
        k_attn<<<NB * NH_ * TT, 256, 0, stream>>>(qkvb, attn);
        launch_gemm(stream, attn, TL_, p_toutw + (size_t)i * TL_ * TL_, TL_,
                    p_toutb + (size_t)i * TL_, ht, TL_, zbuf, TL_,
                    RT_, TL_, TL_, 0);                                // z = ht + a
        k_ln<<<RT_, 256, 0, stream>>>(zbuf, p_ln2s + (size_t)i * TL_,
                                      p_ln2b + (size_t)i * TL_, nullptr, znbuf, TL_, TT);
        launch_gemm(stream, znbuf, TL_, p_w1 + (size_t)i * TL_ * FFN_, FFN_,
                    p_b1 + (size_t)i * FFN_, nullptr, 0, f1, FFN_,
                    RT_, FFN_, TL_, 1);                               // gelu
        launch_gemm(stream, f1, FFN_, p_w2 + (size_t)i * FFN_ * TL_, TL_,
                    p_b2 + (size_t)i * TL_, zbuf, TL_, ht, TL_,
                    RT_, TL_, FFN_, 0);                               // ht = z + f

        const float* skip = (i >= 4) ? saves[7 - i] : nullptr;
        k_add2<<<GH, 256, 0, stream>>>(ht, skip, hbuf, HELEMS);
    }

    // ---- final RMS + reverse gather + LM head ----
    k_rms<<<RT_, 256, 0, stream>>>(hbuf, p_normf, nullptr, ybuf, TL_, 1);
    k_head<<<(NB * TT * 16 * 3 + 255) / 256, 256, 0, stream>>>(
        ybuf, p_lmw, p_lmb, (float*)d_out);
}
